// BiRNNCRF_6124623364569
// MI455X (gfx1250) — matrix operand order repaired and checked by reading, not compiled, not run on hardware
//
#include <hip/hip_runtime.h>
#include <hip/hip_bf16.h>

#define B_   64
#define T_   1024
#define V_   8002
#define D_   256
#define U_   256
#define K_   5
#define G3   768           // 3*U
#define MTOT 65536         // B*T

typedef __bf16 bf16_t;
typedef __attribute__((ext_vector_type(16))) __bf16 v16bf;
typedef __attribute__((ext_vector_type(8)))  __bf16 v8bf;
typedef __attribute__((ext_vector_type(8)))  float  v8f;
typedef __attribute__((ext_vector_type(4)))  unsigned int u32x4;
typedef __attribute__((ext_vector_type(8)))  int i32x8;
typedef __attribute__((ext_vector_type(4)))  int i32x4;

__device__ __forceinline__ unsigned short f2bf(float f) {
  unsigned int u = __float_as_uint(f);
  u += 0x7FFFu + ((u >> 16) & 1u);          // round-to-nearest-even
  return (unsigned short)(u >> 16);
}
__device__ __forceinline__ float bf2f(unsigned short h) {
  return __uint_as_float(((unsigned int)h) << 16);
}
// A fragment (16x32 bf16): lane holds row M=lane%16; lanes<16 get K 0..15,
// lanes>=16 get 16..31.  p points at this lane's first 8 bf16; 2nd chunk at +8.
__device__ __forceinline__ v16bf make_a_frag(const bf16_t* p) {
  v8bf a0 = *(const v8bf*)p;
  v8bf a1 = *(const v8bf*)(p + 8);
  v16bf a;
#pragma unroll
  for (int i = 0; i < 8; ++i) { a[i] = a0[i]; a[i + 8] = a1[i]; }
  return a;
}
__device__ __forceinline__ v8f wmma_bf16(v16bf a, v16bf b, v8f c) {
  return __builtin_amdgcn_wmma_f32_16x16x32_bf16(false, a, false, b, (short)0, c,
                                                 false, false);
}
__device__ __forceinline__ float sigmoidf_(float x) { return 1.f / (1.f + __expf(-x)); }

// ---------------------------------------------------------------------------
// TDM: async 2D bf16 tile load (tile_dim0=768 cols x `rows` rows) into LDS.
// D# group0: count=1, lds_addr, 57b global addr, type=2.  group1: data_size=2B,
// tensor_dim0=768, tensor_dim1=rows, tile_dim0=768, tile_dim1=rows,
// tensor_dim0_stride=T*768 (elements).  Groups 2/3 zero (2D tensor).
// ---------------------------------------------------------------------------
__device__ __forceinline__ void tdm_load_xg_tile(unsigned lds_off,
                                                 unsigned long long gaddr,
                                                 int rows) {
  u32x4 g0;
  g0[0] = 1u;                                              // count = 1
  g0[1] = lds_off;                                         // LDS byte address
  g0[2] = (unsigned)(gaddr & 0xFFFFFFFFull);               // global_addr[31:0]
  g0[3] = (unsigned)((gaddr >> 32) & 0x1FFFFFFull) | 0x80000000u;  // [56:32] | type=2
  i32x8 g1;
  g1[0] = (int)(1u << 16);                                 // data_size = 1 -> 2 bytes
  g1[1] = (int)((unsigned)G3 << 16);                       // tensor_dim0[15:0]
  g1[2] = (int)((unsigned)rows << 16);                     // tensor_dim1[15:0]
  g1[3] = (int)((unsigned)G3 << 16);                       // tile_dim0
  g1[4] = rows;                                            // tile_dim1 (tile_dim2=0)
  g1[5] = (int)(T_ * G3);                                  // tensor_dim0_stride[31:0]
  g1[6] = 0;
  g1[7] = 0;
  i32x4 z4 = {0, 0, 0, 0};
#if __clang_major__ >= 23
  i32x8 z8 = {0, 0, 0, 0, 0, 0, 0, 0};
  __builtin_amdgcn_tensor_load_to_lds(g0, g1, z4, z4, z8, 0);
#else
  __builtin_amdgcn_tensor_load_to_lds(g0, g1, z4, z4, 0);
#endif
}

// ---------------------------------------------------------------------------
// Kernel 1: fp32 -> bf16 staging. emb row-major; Wx/Wh transposed to [N][K]
// so WMMA B-fragments are single contiguous 32B loads per lane.
// ---------------------------------------------------------------------------
#define E_EMB  (V_ * D_)                 // 2,048,512
#define E_WX   (2 * G3 * U_)             //   393,216  ([1536][256], fwd cols then bwd)
#define E_WH   (G3 * U_)                 //   196,608 per direction
#define E_TOT  (E_EMB + E_WX + 2 * E_WH)

__global__ __launch_bounds__(256) void prep_kernel(
    const float* __restrict__ emb,
    const float* __restrict__ Wxf, const float* __restrict__ Wxb,
    const float* __restrict__ Whf, const float* __restrict__ Whb,
    unsigned short* __restrict__ embb, unsigned short* __restrict__ wxT,
    unsigned short* __restrict__ whTf, unsigned short* __restrict__ whTb) {
  int idx = blockIdx.x * blockDim.x + threadIdx.x;
  if (idx >= E_TOT) return;
  if (idx < E_EMB) {
    embb[idx] = f2bf(emb[idx]);
  } else if (idx < E_EMB + E_WX) {
    int i = idx - E_EMB;
    int n = i >> 8, k = i & 255;                      // [n][k]
    float v = (n < G3) ? Wxf[k * G3 + n] : Wxb[k * G3 + (n - G3)];
    wxT[i] = f2bf(v);
  } else if (idx < E_EMB + E_WX + E_WH) {
    int i = idx - E_EMB - E_WX;
    int n = i >> 8, k = i & 255;
    whTf[i] = f2bf(Whf[k * G3 + n]);
  } else {
    int i = idx - E_EMB - E_WX - E_WH;
    int n = i >> 8, k = i & 255;
    whTb[i] = f2bf(Whb[k * G3 + n]);
  }
}

// ---------------------------------------------------------------------------
// Kernel 2: fused embedding gather + xg = x@Wx + b0, both directions (N=1536).
// Grid (512, 12); 8 waves/WG; wave w: 16 gathered rows x 8 N-tiles (128 cols).
// ---------------------------------------------------------------------------
__global__ __launch_bounds__(256) void xg_gemm_kernel(
    const int* __restrict__ tokens, const unsigned short* __restrict__ embb,
    const unsigned short* __restrict__ wxT,
    const float* __restrict__ bf0, const float* __restrict__ bb0,
    unsigned short* __restrict__ xgf, unsigned short* __restrict__ xgb) {
  const int wave = threadIdx.x >> 5, lane = threadIdx.x & 31;
  const int ln = lane & 15, hi = lane >> 4;
  const int mbase = blockIdx.x * 128 + wave * 16;
  const int nbase = blockIdx.y * 128;

  const int tok = tokens[mbase + ln];                 // row gathered per lane
  const bf16_t* arow = (const bf16_t*)embb + (long)tok * D_;
  const bf16_t* wt   = (const bf16_t*)wxT;

  v8f acc[8] = {};
#pragma unroll
  for (int c = 0; c < 8; ++c) {                       // K = 256 in 8 chunks of 32
    v16bf a = make_a_frag(arow + c * 32 + hi * 16);
#pragma unroll
    for (int n = 0; n < 8; ++n) {
      const bf16_t* bp = wt + (long)(nbase + n * 16 + ln) * U_ + c * 32 + hi * 16;
      acc[n] = wmma_bf16(a, *(const v16bf*)bp, acc[n]);
    }
  }
#pragma unroll
  for (int n = 0; n < 8; ++n) {
    const int col = nbase + n * 16 + ln;
    const float bias = (col < G3) ? bf0[col] : bb0[col - G3];
#pragma unroll
    for (int r = 0; r < 8; ++r) {
      const long m = mbase + r + 8 * hi;
      const unsigned short v = f2bf(acc[n][r] + bias);
      if (col < G3) xgf[m * G3 + col]        = v;
      else          xgb[m * G3 + (col - G3)] = v;
    }
  }
}

// ---------------------------------------------------------------------------
// Kernel 3: GRU scan. Grid (4, 2): x = batch group of 16, y = direction.
// 16 waves; wave w owns u-columns [16w,16w+16) of all three gates so the
// elementwise update runs straight out of its own accumulators; h lives in
// registers; bf16 h mirror in LDS for A-fragments; Wh streamed from L2.
// The next step's xg tile (16x768 bf16, 24KB) is DMA'd into a double-buffered
// LDS tile by the Tensor Data Mover, overlapped with the WMMA phase.
// ---------------------------------------------------------------------------
#define HB_STRIDE 264          // padded bf16 h tile row to dodge LDS bank conflicts
#define XGTILE_BYTES (16 * G3 * 2)                 // 24576
#define HBF_OFF      (2 * XGTILE_BYTES)            // 49152
#define SCAN_LDS_BYTES (HBF_OFF + 16 * HB_STRIDE * 2)   // 57600

__global__ __launch_bounds__(512) void gru_scan_kernel(
    const unsigned short* __restrict__ xgf, const unsigned short* __restrict__ xgb,
    const unsigned short* __restrict__ whTf, const unsigned short* __restrict__ whTb,
    const float* __restrict__ bf1, const float* __restrict__ bb1,
    float* __restrict__ hf, float* __restrict__ hb) {
  extern __shared__ char smem[];
  unsigned short* hbf = (unsigned short*)(smem + HBF_OFF);

  const int dir = blockIdx.y;
  const unsigned short* xg  = dir ? xgb : xgf;
  const bf16_t*         whT = (const bf16_t*)(dir ? whTb : whTf);
  const float*          b1  = dir ? bb1 : bf1;
  float*                ho  = dir ? hb : hf;

  const int wave = threadIdx.x >> 5, lane = threadIdx.x & 31;
  const int ln = lane & 15, hi = lane >> 4;
  const int b0 = blockIdx.x * 16;
  const int ucol = wave * 16 + ln;                    // this lane's u column

  for (int i = threadIdx.x; i < 16 * HB_STRIDE; i += blockDim.x) hbf[i] = 0;

  const unsigned lds_dyn = __builtin_amdgcn_groupstaticsize();
  const unsigned long long xg_base = (unsigned long long)xg;

  // stage tile for the first step
  if (wave == 0) {
    const int t0 = dir ? (T_ - 1) : 0;
    tdm_load_xg_tile(lds_dyn,
                     xg_base + (unsigned long long)((unsigned)(b0 * T_ + t0)) * (G3 * 2),
                     16);
    __builtin_amdgcn_s_wait_tensorcnt(0);
  }
  __syncthreads();

  float hreg[8];
#pragma unroll
  for (int r = 0; r < 8; ++r) hreg[r] = 0.f;

  const float b1z = b1[ucol], b1r = b1[U_ + ucol], b1h = b1[2 * U_ + ucol];
  // B-fragment row pointers for the 3 gates (WhT is [768][256])
  const bf16_t* brz = whT + (long)(0 * U_ + ucol) * U_ + hi * 16;
  const bf16_t* brr = whT + (long)(1 * U_ + ucol) * U_ + hi * 16;
  const bf16_t* brh = whT + (long)(2 * U_ + ucol) * U_ + hi * 16;

  for (int s = 0; s < T_; ++s) {
    const int t = dir ? (T_ - 1 - s) : s;
    // kick off DMA of next step's xg tile; overlapped with the WMMAs below
    if (wave == 0 && (s + 1) < T_) {
      const int tn = dir ? (T_ - 2 - s) : (s + 1);
      tdm_load_xg_tile(lds_dyn + ((s + 1) & 1) * XGTILE_BYTES,
                       xg_base + (unsigned long long)((unsigned)(b0 * T_ + tn)) * (G3 * 2),
                       16);
    }
    v8f az = {}, ar = {}, ah = {};
#pragma unroll
    for (int c = 0; c < 8; ++c) {                     // rg = h @ Wh, K=256
      const bf16_t* ap = (const bf16_t*)&hbf[ln * HB_STRIDE + c * 32 + hi * 16];
      v16bf a = make_a_frag(ap);
      az = wmma_bf16(a, *(const v16bf*)(brz + c * 32), az);
      ar = wmma_bf16(a, *(const v16bf*)(brr + c * 32), ar);
      ah = wmma_bf16(a, *(const v16bf*)(brh + c * 32), ah);
    }
    if (wave == 0) __builtin_amdgcn_s_wait_tensorcnt(0);  // this step's tile landed
    __syncthreads();                                  // all waves done reading h
    const unsigned short* xt = (const unsigned short*)(smem + (s & 1) * XGTILE_BYTES);
#pragma unroll
    for (int r = 0; r < 8; ++r) {
      const int m = r + 8 * hi;                       // row within batch group
      const float xz  = bf2f(xt[m * G3 + ucol]);
      const float xr_ = bf2f(xt[m * G3 + U_ + ucol]);
      const float xh  = bf2f(xt[m * G3 + 2 * U_ + ucol]);
      const float z  = sigmoidf_(xz  + az[r] + b1z);
      const float rr = sigmoidf_(xr_ + ar[r] + b1r);
      const float hh = tanhf(xh + rr * (ah[r] + b1h));
      const float hn = z * hreg[r] + (1.f - z) * hh;
      hreg[r] = hn;
      hbf[m * HB_STRIDE + ucol] = f2bf(hn);
      ho[((long)(b0 + m) * T_ + t) * U_ + ucol] = hn;
    }
    __syncthreads();                                  // h updated for next step
  }
}

// ---------------------------------------------------------------------------
// Kernel 4: h = 0.5(hf+hb); emit = softmax(h@Wd + bd).  One wave per row.
// ---------------------------------------------------------------------------
__global__ __launch_bounds__(256) void dense_softmax_kernel(
    const float* __restrict__ hf, const float* __restrict__ hb,
    const float* __restrict__ Wd, const float* __restrict__ bd,
    float* __restrict__ emit) {
  const int wave = threadIdx.x >> 5, lane = threadIdx.x & 31;
  const long row = (long)blockIdx.x * 8 + wave;
  const float* pf = hf + row * U_;
  const float* pb = hb + row * U_;
  float acc[K_] = {0.f, 0.f, 0.f, 0.f, 0.f};
  for (int u = lane; u < U_; u += 32) {
    const float h = 0.5f * (pf[u] + pb[u]);
#pragma unroll
    for (int k = 0; k < K_; ++k) acc[k] += h * Wd[u * K_ + k];
  }
#pragma unroll
  for (int k = 0; k < K_; ++k)
#pragma unroll
    for (int off = 16; off > 0; off >>= 1) acc[k] += __shfl_xor(acc[k], off);
  if (lane == 0) {
    float lg[K_], m = -1e30f, s = 0.f;
#pragma unroll
    for (int k = 0; k < K_; ++k) { lg[k] = acc[k] + bd[k]; m = fmaxf(m, lg[k]); }
#pragma unroll
    for (int k = 0; k < K_; ++k) { lg[k] = __expf(lg[k] - m); s += lg[k]; }
    const float inv = 1.f / s;
#pragma unroll
    for (int k = 0; k < K_; ++k) emit[row * K_ + k] = lg[k] * inv;
  }
}

// ---------------------------------------------------------------------------
// Kernel 5: CRF forward-backward. One wave per batch element; lane j < 5
// carries tag j; shfl broadcasts the 5-vector each step.
// ---------------------------------------------------------------------------
__global__ __launch_bounds__(256) void crf_kernel(
    const float* __restrict__ emit, const float* __restrict__ trans,
    float* __restrict__ alpha, float* __restrict__ out) {
  const int wave = threadIdx.x >> 5, lane = threadIdx.x & 31;
  const int b = blockIdx.x * 8 + wave;
  const int j = (lane < K_) ? lane : (K_ - 1);        // clamp; lanes>=5 shadow tag 4
  float tcol[K_], trow[K_];
#pragma unroll
  for (int i = 0; i < K_; ++i) { tcol[i] = trans[i * K_ + j]; trow[i] = trans[j * K_ + i]; }
  const float* eb = emit + (long)b * T_ * K_;
  float* ab = alpha + (long)b * T_ * K_;
  float* ob = out + (long)b * T_ * K_;

  // forward
  float a = eb[j];
  if (lane < K_) ab[j] = a;
  for (int t = 1; t < T_; ++t) {
    float av[K_];
#pragma unroll
    for (int i = 0; i < K_; ++i) av[i] = __shfl(a, i);
    float m = av[0] + tcol[0];
#pragma unroll
    for (int i = 1; i < K_; ++i) m = fmaxf(m, av[i] + tcol[i]);
    float s = 0.f;
#pragma unroll
    for (int i = 0; i < K_; ++i) s += __expf(av[i] + tcol[i] - m);
    a = m + __logf(s) + eb[t * K_ + j];
    if (lane < K_) ab[t * K_ + j] = a;
  }
  // backward + output
  float bv = 0.f;
  {   // t = T-1: beta = 0, a already holds alpha[T-1]
    float x = a, xv[K_];
#pragma unroll
    for (int i = 0; i < K_; ++i) xv[i] = __shfl(x, i);
    float m = xv[0];
#pragma unroll
    for (int i = 1; i < K_; ++i) m = fmaxf(m, xv[i]);
    float s = 0.f;
#pragma unroll
    for (int i = 0; i < K_; ++i) s += __expf(xv[i] - m);
    if (lane < K_) ob[(T_ - 1) * K_ + j] = __expf(x - m) / s;
  }
  for (int t = T_ - 2; t >= 0; --t) {
    float c = eb[(t + 1) * K_ + j] + bv;
    float cv[K_];
#pragma unroll
    for (int i = 0; i < K_; ++i) cv[i] = __shfl(c, i);
    float m = trow[0] + cv[0];
#pragma unroll
    for (int i = 1; i < K_; ++i) m = fmaxf(m, trow[i] + cv[i]);
    float s = 0.f;
#pragma unroll
    for (int i = 0; i < K_; ++i) s += __expf(trow[i] + cv[i] - m);
    bv = m + __logf(s);
    const float x = ab[t * K_ + j] + bv;
    float xv[K_];
#pragma unroll
    for (int i = 0; i < K_; ++i) xv[i] = __shfl(x, i);
    float mm = xv[0];
#pragma unroll
    for (int i = 1; i < K_; ++i) mm = fmaxf(mm, xv[i]);
    float ss = 0.f;
#pragma unroll
    for (int i = 0; i < K_; ++i) ss += __expf(xv[i] - mm);
    if (lane < K_) ob[t * K_ + j] = __expf(x - mm) / ss;
  }
}

// ---------------------------------------------------------------------------
extern "C" void kernel_launch(void* const* d_in, const int* in_sizes, int n_in,
                              void* d_out, int out_size, void* d_ws, size_t ws_size,
                              hipStream_t stream) {
  (void)in_sizes; (void)n_in; (void)out_size; (void)ws_size;
  const int*   tokens = (const int*)  d_in[0];
  const float* emb    = (const float*)d_in[1];
  const float* Wx_f   = (const float*)d_in[2];
  const float* Wh_f   = (const float*)d_in[3];
  const float* b_f    = (const float*)d_in[4];   // [2][768]
  const float* Wx_b   = (const float*)d_in[5];
  const float* Wh_b   = (const float*)d_in[6];
  const float* b_b    = (const float*)d_in[7];
  const float* Wd     = (const float*)d_in[8];
  const float* bd     = (const float*)d_in[9];
  const float* trans  = (const float*)d_in[10];
  float* out = (float*)d_out;

  char* ws = (char*)d_ws;
  size_t off = 0;
  auto alloc = [&](size_t bytes) -> void* {
    void* p = ws + off;
    off = (off + bytes + 255) & ~(size_t)255;
    return p;
  };
  unsigned short* embb = (unsigned short*)alloc((size_t)E_EMB * 2);
  unsigned short* wxT  = (unsigned short*)alloc((size_t)E_WX * 2);
  unsigned short* whTf = (unsigned short*)alloc((size_t)E_WH * 2);
  unsigned short* whTb = (unsigned short*)alloc((size_t)E_WH * 2);
  unsigned short* xgf  = (unsigned short*)alloc((size_t)MTOT * G3 * 2);
  unsigned short* xgb  = (unsigned short*)alloc((size_t)MTOT * G3 * 2);
  float* hf    = (float*)alloc((size_t)MTOT * U_ * 4);
  float* hb    = (float*)alloc((size_t)MTOT * U_ * 4);
  float* emit  = (float*)alloc((size_t)MTOT * K_ * 4);
  float* alpha = (float*)alloc((size_t)MTOT * K_ * 4);

  prep_kernel<<<(E_TOT + 255) / 256, 256, 0, stream>>>(
      emb, Wx_f, Wx_b, Wh_f, Wh_b, embb, wxT, whTf, whTb);

  xg_gemm_kernel<<<dim3(MTOT / 128, (2 * G3) / 128), 256, 0, stream>>>(
      tokens, embb, wxT, b_f, b_b, xgf, xgb);

  gru_scan_kernel<<<dim3(B_ / 16, 2), 512, SCAN_LDS_BYTES, stream>>>(
      xgf, xgb, whTf, whTb, b_f + G3, b_b + G3, hf, hb);

  dense_softmax_kernel<<<MTOT / 8, 256, 0, stream>>>(hf, hb, Wd, bd, emit);

  crf_kernel<<<B_ / 8, 256, 0, stream>>>(emit, trans, alpha, out);
}
